// TreeLSTM_layer_dgl_36215164240833
// MI455X (gfx1250) — matrix operand order repaired and checked by reading, not compiled, not run on hardware
//
#include <hip/hip_runtime.h>

// ---------------- problem constants (from reference) ----------------
#define DEPTH     13
#define N_NODES   ((1 << DEPTH) - 1)        // 8191
#define N_EDGES   (N_NODES - 1)             // 8190
#define MAX_LEN   (N_NODES + N_EDGES)       // 16381
#define BATCH     8
#define D_MODEL   256
#define M_TOTAL   (MAX_LEN * BATCH)         // 131048 rows of the big GEMM

// LDS row strides produced by TDM padding: +16B after every 512B
#define LDS_ROW_LIN   528                   // 256 bf16 (512B) + 16B pad  -> 132 dw stride
#define LDS_ROW_MSG   1584                  // 768 bf16 (1536B) + 3*16B  -> 396 dw stride
#define LDS_LIN_BYTES (256 * LDS_ROW_LIN)   // 135168 B
#define LDS_MSG_BYTES (128 * LDS_ROW_MSG)   // 202752 B (half of N)

typedef __bf16 v16bf __attribute__((ext_vector_type(16)));
typedef __bf16 v8bf  __attribute__((ext_vector_type(8)));
typedef float  v8f   __attribute__((ext_vector_type(8)));
typedef unsigned int u32x4 __attribute__((ext_vector_type(4)));
typedef int          i32x4 __attribute__((ext_vector_type(4)));
typedef int          i32x8 __attribute__((ext_vector_type(8)));

// ============================================================================
// TDM: async 2D bf16 tile load, global -> LDS, with 4-DWORD pad every 128 DWORDs
// (D# packing per CDNA5 ISA §8.3/8.4; tile == tensor; groups 2/3 disabled)
// 6-arg builtin form (this toolchain): (g0, g1, g2, g3, g_extra, cpol)
// ============================================================================
__device__ __forceinline__ void tdm_load_2d_bf16(const void* gptr, unsigned lds_byte,
                                                 unsigned width_elems, unsigned height,
                                                 unsigned stride_elems) {
    unsigned long long ga = (unsigned long long)gptr;
    u32x4 g0;
    g0[0] = 1u;                                                   // count=1, user D#
    g0[1] = lds_byte;                                             // lds_addr
    g0[2] = (unsigned)ga;                                         // global_addr[31:0]
    g0[3] = ((unsigned)(ga >> 32) & 0x01FFFFFFu) | 0x80000000u;   // addr[56:32] | type=2
    i32x8 g1;
    g1[0] = (int)((1u << 16)        // data_size = 1 -> 2 bytes
                | (1u << 20)        // pad_enable
                | (6u << 22)        // pad_interval: 128 DWORDs (512B)
                | (3u << 25));      // pad_amount:   4 DWORDs (16B)
    g1[1] = (int)(width_elems << 16);                             // tensor_dim0[15:0]
    g1[2] = (int)((width_elems >> 16) | (height << 16));          // dim0 hi | tensor_dim1 lo
    g1[3] = (int)((height >> 16) | (width_elems << 16));          // dim1 hi | tile_dim0
    g1[4] = (int)height;                                          // tile_dim1 | tile_dim2=0
    g1[5] = (int)stride_elems;                                    // tensor_dim0_stride lo
    g1[6] = 0;                                                    // stride0 hi | stride1 lo
    g1[7] = 0;
    i32x4 z4 = (i32x4){0, 0, 0, 0};
    i32x8 z8 = (i32x8){0, 0, 0, 0, 0, 0, 0, 0};
    __builtin_amdgcn_tensor_load_to_lds(g0, g1, z4, z4, z8, 0);
}

// A fragment, 16x32 bf16: lane holds row M=lane%16; one contiguous 16-float run (k = 16*h + s),
// converted with v_cvt_pk_bf16_f32 (RNE) by the backend.
__device__ __forceinline__ v16bf load_a_frag_f32(const float* __restrict__ row,
                                                 int k0, int lane_hi) {
    const float4* p0 = (const float4*)(row + k0 + lane_hi * 16);
    const float4* p1 = (const float4*)(row + k0 + 8 + lane_hi * 16);
    float4 r0 = p0[0], r1 = p0[1], r2 = p1[0], r3 = p1[1];
    v16bf a;
    a[0]  = (__bf16)r0.x; a[1]  = (__bf16)r0.y; a[2]  = (__bf16)r0.z; a[3]  = (__bf16)r0.w;
    a[4]  = (__bf16)r1.x; a[5]  = (__bf16)r1.y; a[6]  = (__bf16)r1.z; a[7]  = (__bf16)r1.w;
    a[8]  = (__bf16)r2.x; a[9]  = (__bf16)r2.y; a[10] = (__bf16)r2.z; a[11] = (__bf16)r2.w;
    a[12] = (__bf16)r3.x; a[13] = (__bf16)r3.y; a[14] = (__bf16)r3.z; a[15] = (__bf16)r3.w;
    return a;
}

// B fragment from LDS (bank-conflict-free thanks to the TDM pad): two ds_load_b128
__device__ __forceinline__ v16bf load_b_frag_lds(const char* sm, unsigned off) {
    v8bf lo = *(const v8bf*)(sm + off);
    v8bf hi = *(const v8bf*)(sm + off + 16);
    return __builtin_shufflevector(lo, hi, 0, 1, 2, 3, 4, 5, 6, 7,
                                           8, 9, 10, 11, 12, 13, 14, 15);
}

// ============================================================================
// Kernel 0: per-launch weight conversion f32 -> bf16 into d_ws.
// ============================================================================
__global__ void __launch_bounds__(256)
convert_weights_kernel(const float* __restrict__ Wlin,
                       const float* __restrict__ Wmsg,
                       __bf16* __restrict__ wlin_bf,
                       __bf16* __restrict__ wmsg_bf) {
    const int i = blockIdx.x * blockDim.x + threadIdx.x;
    if (i < D_MODEL * D_MODEL)     wlin_bf[i] = (__bf16)Wlin[i];
    if (i < D_MODEL * 3 * D_MODEL) wmsg_bf[i] = (__bf16)Wmsg[i];
}

// ============================================================================
// Kernel 1: Y[m, n] = sum_k A[m, k] * W_lin[n, k]   (Y = d_out, f32)
// Block = 8 waves x 16-row strips (128 rows). Wave 0 TDM-stages all of W_lin
// bf16 into LDS; B fragments come from ds_load_b128 thereafter.
// ============================================================================
__global__ void __launch_bounds__(256)
gemm_lin_strip_kernel(const float* __restrict__ A,
                      const __bf16* __restrict__ Wb,
                      float* __restrict__ Y) {
    extern __shared__ char smem[];
    const int lane = threadIdx.x & 31;
    const int wave = threadIdx.x >> 5;

    // ---- stage weights (wave 0 issues the TDM op; EXEC ignored by TDM) ----
    if (wave == 0) {
        tdm_load_2d_bf16(Wb, (unsigned)(unsigned long long)(void*)smem,
                         D_MODEL, D_MODEL, D_MODEL);
        __builtin_amdgcn_s_wait_tensorcnt(0);
    }
    __syncthreads();

    const int mt = blockIdx.x * (blockDim.x >> 5) + wave;
    const int MT = (M_TOTAL + 15) / 16;
    if (mt >= MT) return;                     // wave-uniform (after barrier)

    const int m_base  = mt * 16;
    const int lane_lo = lane & 15;
    const int lane_hi = lane >> 4;

    int mrow = m_base + lane_lo;
    if (mrow >= M_TOTAL) mrow = M_TOTAL - 1;  // clamp; masked at store
    const float* Arow = A + (size_t)mrow * D_MODEL;
    __builtin_prefetch(Arow, 0, 0);

    const unsigned blane = (unsigned)(lane_lo * LDS_ROW_LIN + lane_hi * 32);

    v8f acc[16];
    #pragma unroll
    for (int nt = 0; nt < 16; ++nt) acc[nt] = (v8f){};

    #pragma unroll
    for (int k0 = 0; k0 < D_MODEL; k0 += 32) {
        v16bf a = load_a_frag_f32(Arow, k0, lane_hi);
        #pragma unroll
        for (int nt = 0; nt < 16; ++nt) {
            v16bf b = load_b_frag_lds(smem, blane + (unsigned)(nt * 16 * LDS_ROW_LIN + k0 * 2));
            acc[nt] = __builtin_amdgcn_wmma_f32_16x16x32_bf16(
                          false, a, false, b, (short)0, acc[nt], false, false);
        }
    }

    #pragma unroll
    for (int nt = 0; nt < 16; ++nt) {
        #pragma unroll
        for (int r = 0; r < 8; ++r) {
            int m = m_base + r + lane_hi * 8;
            if (m < M_TOTAL)
                Y[(size_t)m * D_MODEL + nt * 16 + lane_lo] = acc[nt][r];
        }
    }
}

// ============================================================================
// Kernel 2: one tree level.  Block owns HALF of N (128 cols): TDM-stages a
// 128x768 bf16 W_msg slice (198KB) into LDS; 8 waves x 16 message rows.
// Each A fragment feeds 8 WMMAs; each B fragment is shared by L and R.
// ============================================================================
__global__ void __launch_bounds__(256)
tree_level_strip_kernel(const __bf16* __restrict__ Wmsgb,
                        float* __restrict__ Y,
                        int level_first,          // (1<<d) - 1
                        int rows) {               // (1<<d) * 8 message rows
    extern __shared__ char smem[];
    const int lane = threadIdx.x & 31;
    const int wave = threadIdx.x >> 5;
    const int half = blockIdx.x & 1;              // which 128-column half of N
    const int mtb  = blockIdx.x >> 1;

    // ---- stage this block's weight slice ----
    if (wave == 0) {
        const __bf16* wslice = Wmsgb + (size_t)half * 128 * (3 * D_MODEL);
        tdm_load_2d_bf16(wslice, (unsigned)(unsigned long long)(void*)smem,
                         3 * D_MODEL, 128, 3 * D_MODEL);
        __builtin_amdgcn_s_wait_tensorcnt(0);
    }
    __syncthreads();

    const int mt = mtb * (blockDim.x >> 5) + wave;
    const int MT = (rows + 15) / 16;
    if (mt >= MT) return;                     // wave-uniform (after barrier)

    const int m_base  = mt * 16;
    const int lane_lo = lane & 15;
    const int lane_hi = lane >> 4;

    int mr = m_base + lane_lo;
    if (mr >= rows) mr = rows - 1;            // clamp; masked at store
    const int  p_idx = mr >> 3;
    const int  b     = mr & 7;
    const long p  = (long)level_first + p_idx;
    const long lc = 2 * p + 1;
    const long rc = 2 * p + 2;

    const float* segL0 = Y + ((lc * BATCH) + b) * (size_t)D_MODEL;                       // x[lc]
    const float* segR0 = Y + ((rc * BATCH) + b) * (size_t)D_MODEL;                       // x[rc]
    const float* segP  = Y + ((p  * BATCH) + b) * (size_t)D_MODEL;                       // x[p]
    const float* segL2 = Y + (((long)N_NODES + (lc - 1)) * BATCH + b) * (size_t)D_MODEL; // e[lc-1]
    const float* segR2 = Y + (((long)N_NODES + (rc - 1)) * BATCH + b) * (size_t)D_MODEL; // e[rc-1]

    __builtin_prefetch(segL0, 0, 0);
    __builtin_prefetch(segR0, 0, 0);
    __builtin_prefetch(segP,  0, 0);
    __builtin_prefetch(segL2, 0, 0);
    __builtin_prefetch(segR2, 0, 0);

    const unsigned blane = (unsigned)(lane_lo * LDS_ROW_MSG + lane_hi * 32);

    v8f accL[8], accR[8];
    #pragma unroll
    for (int nt = 0; nt < 8; ++nt) { accL[nt] = (v8f){}; accR[nt] = (v8f){}; }

    for (int k0 = 0; k0 < 3 * D_MODEL; k0 += 32) {
        const int seg = k0 >> 8;              // each 32-wide K step is one segment
        const int kk  = k0 & (D_MODEL - 1);
        const float* aLrow = (seg == 0) ? segL0 : (seg == 1) ? segP : segL2;
        const float* aRrow = (seg == 0) ? segR0 : (seg == 1) ? segP : segR2;

        v16bf aL = load_a_frag_f32(aLrow, kk, lane_hi);
        v16bf aR = load_a_frag_f32(aRrow, kk, lane_hi);

        // LDS byte offset within a row: k*2 plus one 16B pad per 256 elements
        const unsigned koff = (unsigned)(k0 * 2 + (k0 >> 8) * 16);
        #pragma unroll
        for (int nt = 0; nt < 8; ++nt) {
            v16bf bf = load_b_frag_lds(smem, blane + (unsigned)(nt * 16 * LDS_ROW_MSG) + koff);
            accL[nt] = __builtin_amdgcn_wmma_f32_16x16x32_bf16(
                           false, aL, false, bf, (short)0, accL[nt], false, false);
            accR[nt] = __builtin_amdgcn_wmma_f32_16x16x32_bf16(
                           false, aR, false, bf, (short)0, accR[nt], false, false);
        }
    }

    // x[p] = min(m_l, m_r); output row = level_first*8 + message_row
    const size_t out_base = (size_t)level_first * BATCH;
    const int    col_base = half * 128;
    #pragma unroll
    for (int nt = 0; nt < 8; ++nt) {
        #pragma unroll
        for (int r = 0; r < 8; ++r) {
            int m = m_base + r + lane_hi * 8;
            if (m < rows) {
                float v = fminf(accL[nt][r], accR[nt][r]);
                Y[(out_base + m) * D_MODEL + col_base + nt * 16 + lane_lo] = v;
            }
        }
    }
}

// ============================================================================
// Host launcher
// inputs: 0=all_feats f32, 1=W_lin f32, 2=W_msg f32, 3=seg_ids (unused),
//         4=connectivity (unused).  d_out: concat([x, e]) (16381, 8, 256) f32.
// d_ws layout: [0, 128KB) W_lin bf16 ; [128KB, 512KB) W_msg bf16.
// ============================================================================
extern "C" void kernel_launch(void* const* d_in, const int* in_sizes, int n_in,
                              void* d_out, int out_size, void* d_ws, size_t ws_size,
                              hipStream_t stream) {
    const float* all_feats = (const float*)d_in[0];
    const float* W_lin     = (const float*)d_in[1];
    const float* W_msg     = (const float*)d_in[2];
    float* Y = (float*)d_out;

    __bf16* wlin_bf = (__bf16*)d_ws;
    __bf16* wmsg_bf = (__bf16*)((char*)d_ws + (size_t)D_MODEL * D_MODEL * sizeof(__bf16));

    // ---- convert weights to bf16 (512 KB of scratch, lives in L2) ----
    {
        const int n      = D_MODEL * 3 * D_MODEL;
        const int blocks = (n + 255) / 256;
        convert_weights_kernel<<<blocks, 256, 0, stream>>>(W_lin, W_msg, wlin_bf, wmsg_bf);
    }

    // ---- y = all_feats @ W_lin.T ----
    {
        const int blocks = (M_TOTAL + 127) / 128;     // 8 waves x 16 rows per block
        gemm_lin_strip_kernel<<<blocks, 256, LDS_LIN_BYTES, stream>>>(all_feats, wlin_bf, Y);
    }

    // ---- bottom-up tree pass: levels d = DEPTH-2 .. 0, stream-serialized ----
    for (int d = DEPTH - 2; d >= 0; --d) {
        const int level_first = (1 << d) - 1;
        const int rows   = (1 << d) * BATCH;
        const int mtb    = (rows + 127) / 128;        // 128 message rows per block
        const int blocks = mtb * 2;                   // x2 N-halves
        tree_level_strip_kernel<<<blocks, 256, LDS_MSG_BYTES, stream>>>(
            wmsg_bf, Y, level_first, rows);
    }
}
